// GATBlock_45200235823722
// MI455X (gfx1250) — matrix operand order repaired and checked by reading, not compiled, not run on hardware
//
#include <hip/hip_runtime.h>
#include <hip/hip_bf16.h>

// ---------------------------------------------------------------------------
// GAT block for MI455X (gfx1250, wave32).
//   N=100000 nodes, E=1.6M edges, IN=128, H=4 heads, C=32 (H*C=128).
//   h = x@W via v_wmma_f32_16x16x32_bf16 (fp32 accumulate),
//   segment softmax via order-preserving-uint atomicMax + f32 atomicAdd,
//   message scatter: one wave per edge, lanes = channels (L2-resident).
// ---------------------------------------------------------------------------

typedef __attribute__((ext_vector_type(16))) __bf16 v16bf;
typedef __attribute__((ext_vector_type(8)))  float  v8f;

#define NEG_SLOPE 0.2f
#define F_DIM 128           // IN == H*C == 128
#define LDS_PITCH 136       // bf16 elems per transposed-W row (+8 pad vs 64 banks)

// ---- helpers --------------------------------------------------------------

__device__ inline unsigned short bf16_bits(float f) {
  unsigned u = __float_as_uint(f);
  return (unsigned short)((u + 0x7FFFu + ((u >> 16) & 1u)) >> 16); // RNE
}
__device__ inline __bf16 bf16_val(float f) {
  unsigned short s = bf16_bits(f);
  return __builtin_bit_cast(__bf16, s);
}
__device__ inline float leaky(float v) { return v > 0.0f ? v : NEG_SLOPE * v; }

// order-preserving float<->uint encoding so unsigned atomicMax == float max
__device__ inline unsigned enc_f32(float f) {
  unsigned b = __float_as_uint(f);
  return (b & 0x80000000u) ? ~b : (b | 0x80000000u);
}
__device__ inline float dec_f32(unsigned v) {
  unsigned b = (v & 0x80000000u) ? (v ^ 0x80000000u) : ~v;
  return __uint_as_float(b);
}

struct alignas(16) Bchunk { unsigned u[8]; };   // 32B -> v16bf bit_cast carrier

// ---- kernel 0: zero accumulators ------------------------------------------

__global__ void k_init(float* __restrict__ out, unsigned* __restrict__ m_enc,
                       float* __restrict__ denom, int N) {
  int i = blockIdx.x * blockDim.x + threadIdx.x;
  int total = N * F_DIM;
  if (i < total) out[i] = 0.0f;
  if (i < N * 4) { m_enc[i] = 0u; denom[i] = 0.0f; }
}

// ---- kernel 1: h = x @ W  (bf16 WMMA, fp32 accumulate) --------------------
// Block = 256 threads = 8 waves; each wave owns 16 rows x 128 cols of h.
// W is staged TRANSPOSED in LDS as bf16 with padded pitch so each lane's
// B-operand (16 consecutive K values of one column) is a contiguous 32B read.

__global__ __launch_bounds__(256) void k_gemm(const float* __restrict__ x,
                                              const float* __restrict__ W,
                                              float* __restrict__ h, int N) {
  __shared__ unsigned short sWT[F_DIM * LDS_PITCH]; // sWT[c*PITCH + k] = bf16(W[k][c])

  const int tid = threadIdx.x;
  for (int i = tid; i < F_DIM * F_DIM; i += 256) {
    int k = i >> 7, c = i & 127;
    sWT[c * LDS_PITCH + k] = bf16_bits(W[i]);
  }
  __syncthreads();

  const int lane = tid & 31;
  const int wave = tid >> 5;           // 0..7
  const int m16  = lane & 15;          // M row (A/C) or N col (B/C) within tile
  const int hi   = lane >> 4;          // half-wave select
  const int rowBase = blockIdx.x * 128 + wave * 16;

  int grow = rowBase + m16;
  int lrow = grow < N ? grow : (N - 1);          // clamp loads; EXEC stays full
  const float* xr = x + (size_t)lrow * F_DIM;

  // A operand, contiguous K layout matching B: lanes 0-15 hold K {0..15},
  // lanes 16-31 hold K {16..31} (per K-chunk of 32).
  v16bf A[4];
#pragma unroll
  for (int kk = 0; kk < 4; ++kk) {
    const int off = hi * 16;
    const float4* p0 = (const float4*)(xr + kk * 32 + off);
    const float4* p1 = (const float4*)(xr + kk * 32 + 8 + off);
    float4 f0 = p0[0], f1 = p0[1], f2 = p1[0], f3 = p1[1];
    const float fv[16] = {f0.x, f0.y, f0.z, f0.w, f1.x, f1.y, f1.z, f1.w,
                          f2.x, f2.y, f2.z, f2.w, f3.x, f3.y, f3.z, f3.w};
    v16bf a;
#pragma unroll
    for (int i = 0; i < 16; ++i) a[i] = bf16_val(fv[i]);
    A[kk] = a;
  }

#pragma unroll
  for (int n = 0; n < 8; ++n) {                  // 8 column tiles of 16
    v8f acc = {};
    const unsigned short* bp = &sWT[(n * 16 + m16) * LDS_PITCH];
#pragma unroll
    for (int kk = 0; kk < 4; ++kk) {             // K = 128 in chunks of 32
      // B operand: lane holds K rows (hi*16 .. hi*16+15) of this chunk,
      // element e <-> K = kk*32 + hi*16 + e, column = n*16 + m16.
      const unsigned short* q = bp + kk * 32 + hi * 16;
      Bchunk bc = *(const Bchunk*)q;             // two ds_load_b128
      v16bf b = __builtin_bit_cast(v16bf, bc);
      acc = __builtin_amdgcn_wmma_f32_16x16x32_bf16(
          false, A[kk], false, b, (short)0, acc, false, false);
    }
    // C layout: VGPR r -> M = r (lanes 0-15) / 8+r (lanes 16-31), lane -> N
    const int ccol = n * 16 + m16;
    const int rb   = hi * 8;
#pragma unroll
    for (int r = 0; r < 8; ++r) {
      int rr = rowBase + rb + r;
      if (rr < N) h[(size_t)rr * F_DIM + ccol] = acc[r];
    }
  }
}

// ---- kernel 2: a_src/a_dst = <h[n,head,:], att_*[head,:]> -----------------
// One block per node, 128 threads = 4 waves = 4 heads; wave32 shuffle reduce.

__global__ __launch_bounds__(128) void k_att(const float* __restrict__ h,
                                             const float* __restrict__ att_src,
                                             const float* __restrict__ att_dst,
                                             float* __restrict__ a_src,
                                             float* __restrict__ a_dst) {
  const int n = blockIdx.x, t = threadIdx.x;
  float hv = h[(size_t)n * F_DIM + t];
  float vs = hv * att_src[t];
  float vd = hv * att_dst[t];
#pragma unroll
  for (int o = 16; o > 0; o >>= 1) {
    vs += __shfl_down(vs, o, 32);
    vd += __shfl_down(vd, o, 32);
  }
  if ((t & 31) == 0) {
    a_src[n * 4 + (t >> 5)] = vs;
    a_dst[n * 4 + (t >> 5)] = vd;
  }
}

// ---- edge helpers ---------------------------------------------------------

__device__ inline void edge_sd(const int* __restrict__ ei, int et, int E,
                               int& s, int& d) {
  if (et < E) { s = ei[et]; d = ei[E + et]; }   // edge_index[0]=src, [1]=dst
  else        { s = d = et - E; }               // appended self-loops
}

// ---- kernel 3: segment max of leaky-relu logits (encoded atomicMax) -------

__global__ void k_edge_max(const int* __restrict__ ei,
                           const float* __restrict__ a_src,
                           const float* __restrict__ a_dst,
                           unsigned* __restrict__ m_enc, int E, int Et) {
  int et = blockIdx.x * blockDim.x + threadIdx.x;
  if (et >= Et) return;
  int s, d; edge_sd(ei, et, E, s, d);
  float4 as = *(const float4*)(a_src + (size_t)s * 4);
  float4 ad = *(const float4*)(a_dst + (size_t)d * 4);
  atomicMax(&m_enc[d * 4 + 0], enc_f32(leaky(as.x + ad.x)));
  atomicMax(&m_enc[d * 4 + 1], enc_f32(leaky(as.y + ad.y)));
  atomicMax(&m_enc[d * 4 + 2], enc_f32(leaky(as.z + ad.z)));
  atomicMax(&m_enc[d * 4 + 3], enc_f32(leaky(as.w + ad.w)));
}

// ---- kernel 4: softmax denominator ----------------------------------------

__global__ void k_edge_denom(const int* __restrict__ ei,
                             const float* __restrict__ a_src,
                             const float* __restrict__ a_dst,
                             const unsigned* __restrict__ m_enc,
                             float* __restrict__ denom, int E, int Et) {
  int et = blockIdx.x * blockDim.x + threadIdx.x;
  if (et >= Et) return;
  int s, d; edge_sd(ei, et, E, s, d);
  float4 as = *(const float4*)(a_src + (size_t)s * 4);
  float4 ad = *(const float4*)(a_dst + (size_t)d * 4);
  uint4  me = *(const uint4*)(m_enc + (size_t)d * 4);
  atomicAdd(&denom[d * 4 + 0], __expf(leaky(as.x + ad.x) - dec_f32(me.x)));
  atomicAdd(&denom[d * 4 + 1], __expf(leaky(as.y + ad.y) - dec_f32(me.y)));
  atomicAdd(&denom[d * 4 + 2], __expf(leaky(as.z + ad.z) - dec_f32(me.z)));
  atomicAdd(&denom[d * 4 + 3], __expf(leaky(as.w + ad.w) - dec_f32(me.w)));
}

// ---- kernel 5: weighted scatter  out[dst] += alpha * h[src] ---------------
// One wave per edge; lane = channel; 4 heads -> 4 coalesced 128B segments.

__global__ __launch_bounds__(256) void k_scatter(const float* __restrict__ h,
                                                 const int* __restrict__ ei,
                                                 const float* __restrict__ a_src,
                                                 const float* __restrict__ a_dst,
                                                 const unsigned* __restrict__ m_enc,
                                                 const float* __restrict__ denom,
                                                 float* __restrict__ out,
                                                 int E, int Et) {
  int edge = blockIdx.x * (blockDim.x >> 5) + (threadIdx.x >> 5);
  if (edge >= Et) return;                    // wave-uniform exit
  const int lane = threadIdx.x & 31;
  int s, d; edge_sd(ei, edge, E, s, d);

  float4 as = *(const float4*)(a_src + (size_t)s * 4);
  float4 ad = *(const float4*)(a_dst + (size_t)d * 4);
  uint4  me = *(const uint4*)(m_enc + (size_t)d * 4);
  float4 dn = *(const float4*)(denom + (size_t)d * 4);

  float alpha[4];
  alpha[0] = __expf(leaky(as.x + ad.x) - dec_f32(me.x)) / dn.x;
  alpha[1] = __expf(leaky(as.y + ad.y) - dec_f32(me.y)) / dn.y;
  alpha[2] = __expf(leaky(as.z + ad.z) - dec_f32(me.z)) / dn.z;
  alpha[3] = __expf(leaky(as.w + ad.w) - dec_f32(me.w)) / dn.w;

  const float* hs = h + (size_t)s * F_DIM;
  float*       od = out + (size_t)d * F_DIM;
#pragma unroll
  for (int k = 0; k < 4; ++k)                // head k, channel = lane
    atomicAdd(&od[k * 32 + lane], hs[k * 32 + lane] * alpha[k]);
}

// ---- kernel 6: bias + ELU (in place on d_out) -----------------------------

__global__ void k_bias_elu(float* __restrict__ out, const float* __restrict__ bias,
                           int total) {
  int i = blockIdx.x * blockDim.x + threadIdx.x;
  if (i >= total) return;
  float v = out[i] + bias[i & 127];
  out[i] = v > 0.0f ? v : expm1f(v);
}

// ---- launch ---------------------------------------------------------------

extern "C" void kernel_launch(void* const* d_in, const int* in_sizes, int n_in,
                              void* d_out, int out_size, void* d_ws, size_t ws_size,
                              hipStream_t stream) {
  (void)n_in; (void)out_size; (void)ws_size;
  const float* x       = (const float*)d_in[0];
  const int*   ei      = (const int*)d_in[1];   // int32 per harness convention
  const float* W       = (const float*)d_in[2];
  const float* att_src = (const float*)d_in[3];
  const float* att_dst = (const float*)d_in[4];
  const float* bias    = (const float*)d_in[5];

  const int N  = in_sizes[0] / F_DIM;   // 100000
  const int E  = in_sizes[1] / 2;       // 1600000
  const int Et = E + N;                 // with self loops

  // workspace carve-up (all 16B aligned): h | a_src | a_dst | m_enc | denom
  float*    h     = (float*)d_ws;
  float*    a_src = h + (size_t)N * F_DIM;
  float*    a_dst = a_src + (size_t)N * 4;
  unsigned* m_enc = (unsigned*)(a_dst + (size_t)N * 4);
  float*    denom = (float*)(m_enc + (size_t)N * 4);
  float*    out   = (float*)d_out;

  const int total = N * F_DIM;

  k_init<<<(total + 255) / 256, 256, 0, stream>>>(out, m_enc, denom, N);
  k_gemm<<<(N + 127) / 128, 256, 0, stream>>>(x, W, h, N);
  k_att<<<N, 128, 0, stream>>>(h, att_src, att_dst, a_src, a_dst);
  k_edge_max<<<(Et + 255) / 256, 256, 0, stream>>>(ei, a_src, a_dst, m_enc, E, Et);
  k_edge_denom<<<(Et + 255) / 256, 256, 0, stream>>>(ei, a_src, a_dst, m_enc, denom, E, Et);
  k_scatter<<<(Et + 7) / 8, 256, 0, stream>>>(h, ei, a_src, a_dst, m_enc, denom, out, E, Et);
  k_bias_elu<<<(total + 255) / 256, 256, 0, stream>>>(out, bias, total);
}
